// MultiHeadSelfAttention_69664369541657
// MI455X (gfx1250) — matrix operand order repaired and checked by reading, not compiled, not run on hardware
//
#include <hip/hip_runtime.h>

typedef __attribute__((ext_vector_type(16))) __bf16 v16bf;
typedef __attribute__((ext_vector_type(8)))  __bf16 v8bf;
typedef __attribute__((ext_vector_type(8)))  float  v8f;

union ABu { v16bf v; v8bf h[2]; };

constexpr int BATCH = 2, SEQ = 2048, EMB = 1024, NH = 16, DK = 64;
constexpr int ROWS  = BATCH * SEQ;   // 4096 tokens
constexpr int NQKV  = 3 * EMB;       // 3072

__device__ __forceinline__ void wait_ds0() {
  asm volatile("s_wait_dscnt 0" ::: "memory");
}

__device__ __forceinline__ v8f wmma_bf16(v16bf a, v16bf b, v8f c) {
  return __builtin_amdgcn_wmma_f32_16x16x32_bf16(false, a, false, b,
                                                 (short)0, c, false, false);
}

__device__ __forceinline__ v8f vzero8() {
  v8f z;
#pragma unroll
  for (int i = 0; i < 8; ++i) z[i] = 0.0f;
  return z;
}

// ---------------- pack / transpose kernels ----------------

__global__ void __launch_bounds__(256)
cvt_f32_bf16(const float* __restrict__ src, __bf16* __restrict__ dst, int n) {
  int i = blockIdx.x * blockDim.x + threadIdx.x;
  if (i < n) dst[i] = (__bf16)src[i];
}

// src: [K][N] row-major f32;  dst: [N][K] row-major bf16
__global__ void __launch_bounds__(256)
transpose_to_bf16(const float* __restrict__ src, __bf16* __restrict__ dst,
                  int N, int K) {
  int i = blockIdx.x * blockDim.x + threadIdx.x;
  if (i < N * K) {
    int n = i / K, k = i % K;
    dst[i] = (__bf16)src[(size_t)k * N + n];
  }
}

// ---------------- QKV projection GEMM ----------------
// C[4096,3072] = Xb @ W_qkv + b_qkv, scattered into Q/K/V [B,H,S,Dk] bf16.
// One wave computes a 32x64 tile; double-buffered operand prefetch.
__global__ void __launch_bounds__(256)
qkv_gemm(const __bf16* __restrict__ Ab, const __bf16* __restrict__ Wt,
         const float* __restrict__ bias,
         __bf16* __restrict__ Qb, __bf16* __restrict__ Kb, __bf16* __restrict__ Vb) {
  const int lane = threadIdx.x & 31;
  const int wv   = blockIdx.x * (blockDim.x >> 5) + (threadIdx.x >> 5);
  const int rowTile = wv / (NQKV / 64);
  const int colTile = wv % (NQKV / 64);
  const int m0 = rowTile * 32, n0 = colTile * 64;
  const int lq = lane & 15, half = lane >> 4;

  v8f acc[2][4];
#pragma unroll
  for (int rt = 0; rt < 2; ++rt)
#pragma unroll
    for (int nb = 0; nb < 4; ++nb) acc[rt][nb] = vzero8();

  auto loadA = [&](ABu (&a)[2], int k0) {
#pragma unroll
    for (int rt = 0; rt < 2; ++rt) {
      const __bf16* arow = Ab + (size_t)(m0 + rt * 16 + lq) * EMB + k0;
      a[rt].h[0] = *(const v8bf*)(arow + 8 * half);
      a[rt].h[1] = *(const v8bf*)(arow + 16 + 8 * half);
    }
  };
  auto loadB = [&](ABu (&bb)[4], int k0) {
#pragma unroll
    for (int nb = 0; nb < 4; ++nb) {
      const __bf16* brow = Wt + (size_t)(n0 + nb * 16 + lq) * EMB + k0 + 8 * half;
      bb[nb].h[0] = *(const v8bf*)(brow);
      bb[nb].h[1] = *(const v8bf*)(brow + 16);
    }
  };
  auto domma = [&](ABu (&a)[2], ABu (&bb)[4]) {
#pragma unroll
    for (int nb = 0; nb < 4; ++nb)
#pragma unroll
      for (int rt = 0; rt < 2; ++rt)
        acc[rt][nb] = wmma_bf16(a[rt].v, bb[nb].v, acc[rt][nb]);
  };

  ABu a0[2], b0[4], a1[2], b1[4];
  loadA(a0, 0); loadB(b0, 0);
  for (int k0 = 0; k0 < EMB; k0 += 64) {
    loadA(a1, k0 + 32); loadB(b1, k0 + 32);   // prefetch stage 1
    domma(a0, b0);
    if (k0 + 64 < EMB) { loadA(a0, k0 + 64); loadB(b0, k0 + 64); }  // prefetch stage 0
    domma(a1, b1);
  }

#pragma unroll
  for (int nb = 0; nb < 4; ++nb) {
    const int col  = n0 + nb * 16 + lq;
    const int part = col >> 10;          // 0=Q 1=K 2=V
    const int rem  = col & 1023;
    const int hh   = rem >> 6;
    const int d    = rem & 63;
    const float bv = bias[col];
#pragma unroll
    for (int rt = 0; rt < 2; ++rt)
#pragma unroll
      for (int r = 0; r < 8; ++r) {
        const int row = m0 + rt * 16 + r + 8 * half;   // token index
        const int bb = row >> 11, s = row & 2047;
        float v = acc[rt][nb][r] + bv;
        size_t o = ((size_t)(bb * NH + hh) * SEQ + s) * DK + d;
        if (part == 0)      Qb[o] = (__bf16)(v * 0.125f);  // pre-scale 1/sqrt(Dk)
        else if (part == 1) Kb[o] = (__bf16)v;
        else                Vb[o] = (__bf16)v;
      }
  }
}

// ---------------- flash attention (causal) ----------------
// one wave handles a 32-row q tile of one (b, head); K/V staged in LDS per 32-key step
__global__ void __launch_bounds__(128)
attn_kernel(const __bf16* __restrict__ Qb, const __bf16* __restrict__ Kb,
            const __bf16* __restrict__ Vb, __bf16* __restrict__ attn) {
  __shared__ __bf16 smem[4 * 5632];   // per-wave: K 2048 + Vt 2560 + P 1024
  const int lane = threadIdx.x & 31;
  const int widx = threadIdx.x >> 5;
  const int wv   = blockIdx.x * 4 + widx;
  const int qTile = wv & 63;
  const int head  = (wv >> 6) & 15;
  const int b     = wv >> 10;
  const int q0    = qTile * 32;
  const int lq = lane & 15, half = lane >> 4;

  __bf16* kT = smem + widx * 5632;   // [32][64] row-major
  __bf16* vT = kT + 2048;            // transposed [64][32], row stride 40
  __bf16* pT = vT + 2560;            // [32][32]

  const __bf16* Qbase = Qb + (size_t)(b * NH + head) * SEQ * DK;
  const __bf16* Kbase = Kb + (size_t)(b * NH + head) * SEQ * DK;
  const __bf16* Vbase = Vb + (size_t)(b * NH + head) * SEQ * DK;

  // Q tiles -> A operands: [row subtile][d 0..31 | 32..63]
  ABu aq[2][2];
#pragma unroll
  for (int rt = 0; rt < 2; ++rt) {
    const __bf16* qrow = Qbase + (size_t)(q0 + rt * 16 + lq) * DK;
#pragma unroll
    for (int j = 0; j < 2; ++j) {
      aq[rt][j].h[0] = *(const v8bf*)(qrow + 32 * j + 8 * half);
      aq[rt][j].h[1] = *(const v8bf*)(qrow + 32 * j + 16 + 8 * half);
    }
  }

  v8f o[2][4];
#pragma unroll
  for (int rt = 0; rt < 2; ++rt)
#pragma unroll
    for (int nb = 0; nb < 4; ++nb) o[rt][nb] = vzero8();
  float m_i[2][8], l_i[2][8];
#pragma unroll
  for (int rt = 0; rt < 2; ++rt)
#pragma unroll
    for (int r = 0; r < 8; ++r) { m_i[rt][r] = -1e30f; l_i[rt][r] = 0.0f; }

  for (int k0 = 0; k0 < q0 + 32; k0 += 32) {
    // stage K tile row-major
    {
      const v8bf* kr = (const v8bf*)(Kbase + (size_t)(k0 + lane) * DK);
      v8bf* kd = (v8bf*)(kT + lane * DK);
#pragma unroll
      for (int c = 0; c < 8; ++c) kd[c] = kr[c];
    }
    // stage V transposed: Vt[d][key], row stride 40
    {
      const v8bf* vr = (const v8bf*)(Vbase + (size_t)(k0 + lane) * DK);
      __bf16 tmp[64];
#pragma unroll
      for (int c = 0; c < 8; ++c) ((v8bf*)tmp)[c] = vr[c];
#pragma unroll
      for (int d = 0; d < 64; ++d) vT[d * 40 + lane] = tmp[d];
    }
    wait_ds0();

    // scores + softmax per 16-row subtile
#pragma unroll
    for (int rt = 0; rt < 2; ++rt) {
      v8f sc[2];
#pragma unroll
      for (int t = 0; t < 2; ++t) {
        sc[t] = vzero8();
#pragma unroll
        for (int j = 0; j < 2; ++j) {
          ABu bk;
          const __bf16* kro = kT + (size_t)(16 * t + lq) * DK + 32 * j + 8 * half;
          bk.h[0] = *(const v8bf*)(kro);
          bk.h[1] = *(const v8bf*)(kro + 16);
          sc[t] = wmma_bf16(aq[rt][j].v, bk.v, sc[t]);
        }
      }
#pragma unroll
      for (int r = 0; r < 8; ++r) {
        const int M = r + 8 * half;
        const int qg = q0 + rt * 16 + M;
        float s0 = sc[0][r]; if (k0 + lq > qg)      s0 = -1e30f;
        float s1 = sc[1][r]; if (k0 + 16 + lq > qg) s1 = -1e30f;
        float mx = fmaxf(s0, s1);
#pragma unroll
        for (int msk = 1; msk < 16; msk <<= 1)
          mx = fmaxf(mx, __shfl_xor(mx, msk, 32));
        const float mnew  = fmaxf(m_i[rt][r], mx);
        const float alpha = __expf(m_i[rt][r] - mnew);
        const float p0 = __expf(s0 - mnew);
        const float p1 = __expf(s1 - mnew);
        float rs = p0 + p1;
#pragma unroll
        for (int msk = 1; msk < 16; msk <<= 1)
          rs += __shfl_xor(rs, msk, 32);
        l_i[rt][r] = l_i[rt][r] * alpha + rs;
        m_i[rt][r] = mnew;
#pragma unroll
        for (int nb = 0; nb < 4; ++nb) o[rt][nb][r] *= alpha;
        pT[(rt * 16 + M) * 32 + lq]      = (__bf16)p0;
        pT[(rt * 16 + M) * 32 + 16 + lq] = (__bf16)p1;
      }
    }
    wait_ds0();

    // P (A operands) and V (B operands): O += P @ V, B reused across row subtiles
    ABu ap[2];
#pragma unroll
    for (int rt = 0; rt < 2; ++rt) {
      const __bf16* pr = pT + (size_t)(rt * 16 + lq) * 32;
      ap[rt].h[0] = *(const v8bf*)(pr + 8 * half);
      ap[rt].h[1] = *(const v8bf*)(pr + 16 + 8 * half);
    }
#pragma unroll
    for (int nb = 0; nb < 4; ++nb) {
      ABu bv;
      const __bf16* vro = vT + (size_t)(nb * 16 + lq) * 40 + 8 * half;
      bv.h[0] = *(const v8bf*)(vro);
      bv.h[1] = *(const v8bf*)(vro + 16);
#pragma unroll
      for (int rt = 0; rt < 2; ++rt)
        o[rt][nb] = wmma_bf16(ap[rt].v, bv.v, o[rt][nb]);
    }
    wait_ds0();   // protect LDS tiles before next iteration overwrites
  }

  // normalize and emit attn [B*S, E] bf16 (head-concat layout)
#pragma unroll
  for (int rt = 0; rt < 2; ++rt)
#pragma unroll
    for (int r = 0; r < 8; ++r) {
      const float inv = 1.0f / l_i[rt][r];
      const int row = q0 + rt * 16 + r + 8 * half;
      const size_t base = ((size_t)b * SEQ + row) * EMB + head * DK;
#pragma unroll
      for (int nb = 0; nb < 4; ++nb)
        attn[base + nb * 16 + lq] = (__bf16)(o[rt][nb][r] * inv);
    }
}

// ---------------- output projection GEMM (32x64 tiles, double buffered) ----------------
__global__ void __launch_bounds__(256)
out_gemm(const __bf16* __restrict__ Ab, const __bf16* __restrict__ Wt,
         const float* __restrict__ bias, float* __restrict__ out) {
  const int lane = threadIdx.x & 31;
  const int wv   = blockIdx.x * (blockDim.x >> 5) + (threadIdx.x >> 5);
  const int rowTile = wv / (EMB / 64);
  const int colTile = wv % (EMB / 64);
  const int m0 = rowTile * 32, n0 = colTile * 64;
  const int lq = lane & 15, half = lane >> 4;

  v8f acc[2][4];
#pragma unroll
  for (int rt = 0; rt < 2; ++rt)
#pragma unroll
    for (int nb = 0; nb < 4; ++nb) acc[rt][nb] = vzero8();

  auto loadA = [&](ABu (&a)[2], int k0) {
#pragma unroll
    for (int rt = 0; rt < 2; ++rt) {
      const __bf16* arow = Ab + (size_t)(m0 + rt * 16 + lq) * EMB + k0;
      a[rt].h[0] = *(const v8bf*)(arow + 8 * half);
      a[rt].h[1] = *(const v8bf*)(arow + 16 + 8 * half);
    }
  };
  auto loadB = [&](ABu (&bb)[4], int k0) {
#pragma unroll
    for (int nb = 0; nb < 4; ++nb) {
      const __bf16* brow = Wt + (size_t)(n0 + nb * 16 + lq) * EMB + k0 + 8 * half;
      bb[nb].h[0] = *(const v8bf*)(brow);
      bb[nb].h[1] = *(const v8bf*)(brow + 16);
    }
  };
  auto domma = [&](ABu (&a)[2], ABu (&bb)[4]) {
#pragma unroll
    for (int nb = 0; nb < 4; ++nb)
#pragma unroll
      for (int rt = 0; rt < 2; ++rt)
        acc[rt][nb] = wmma_bf16(a[rt].v, bb[nb].v, acc[rt][nb]);
  };

  ABu a0[2], b0[4], a1[2], b1[4];
  loadA(a0, 0); loadB(b0, 0);
  for (int k0 = 0; k0 < EMB; k0 += 64) {
    loadA(a1, k0 + 32); loadB(b1, k0 + 32);
    domma(a0, b0);
    if (k0 + 64 < EMB) { loadA(a0, k0 + 64); loadB(b0, k0 + 64); }
    domma(a1, b1);
  }

#pragma unroll
  for (int nb = 0; nb < 4; ++nb) {
    const int col = n0 + nb * 16 + lq;
    const float bv = bias[col];
#pragma unroll
    for (int rt = 0; rt < 2; ++rt)
#pragma unroll
      for (int r = 0; r < 8; ++r) {
        const int row = m0 + rt * 16 + r + 8 * half;
        out[(size_t)row * EMB + col] = acc[rt][nb][r] + bv;
      }
  }
}

// ---------------- launch ----------------

extern "C" void kernel_launch(void* const* d_in, const int* in_sizes, int n_in,
                              void* d_out, int out_size, void* d_ws, size_t ws_size,
                              hipStream_t stream) {
  (void)in_sizes; (void)n_in; (void)out_size; (void)ws_size;
  const float* X     = (const float*)d_in[0];
  const float* Wqkv  = (const float*)d_in[1];
  const float* bqkv  = (const float*)d_in[2];
  const float* Wout  = (const float*)d_in[3];
  const float* bout  = (const float*)d_in[4];
  float* out = (float*)d_out;

  char* ws = (char*)d_ws;
  size_t off = 0;
  auto carve = [&](size_t elems) {
    void* p = ws + off;
    off = (off + elems * sizeof(__bf16) + 255) & ~(size_t)255;
    return (__bf16*)p;
  };
  __bf16* Xb     = carve((size_t)ROWS * EMB);
  __bf16* Wqkv_t = carve((size_t)NQKV * EMB);
  __bf16* Wout_t = carve((size_t)EMB * EMB);
  __bf16* Qb     = carve((size_t)BATCH * NH * SEQ * DK);
  __bf16* Kb     = carve((size_t)BATCH * NH * SEQ * DK);
  __bf16* Vb     = carve((size_t)BATCH * NH * SEQ * DK);
  __bf16* attn   = carve((size_t)ROWS * EMB);

  // pack / transpose
  {
    int n = ROWS * EMB;
    cvt_f32_bf16<<<(n + 255) / 256, 256, 0, stream>>>(X, Xb, n);
  }
  {
    int n = NQKV * EMB;
    transpose_to_bf16<<<(n + 255) / 256, 256, 0, stream>>>(Wqkv, Wqkv_t, NQKV, EMB);
  }
  {
    int n = EMB * EMB;
    transpose_to_bf16<<<(n + 255) / 256, 256, 0, stream>>>(Wout, Wout_t, EMB, EMB);
  }

  // QKV projection: (4096/32)*(3072/64) = 6144 waves, 8 per block
  qkv_gemm<<<6144 / 8, 256, 0, stream>>>(Xb, Wqkv_t, bqkv, Qb, Kb, Vb);

  // attention: B*H*(S/32) = 2048 waves, 4 per block (44KB LDS/block)
  attn_kernel<<<2048 / 4, 128, 0, stream>>>(Qb, Kb, Vb, attn);

  // output projection: (4096/32)*(1024/64) = 2048 waves, 8 per block
  out_gemm<<<2048 / 8, 256, 0, stream>>>(attn, Wout_t, bout, out);
}
